// SpanEncoderCrossAttention_88811333747048
// MI455X (gfx1250) — matrix operand order repaired and checked by reading, not compiled, not run on hardware
//
#include <hip/hip_runtime.h>
#include <cstddef>

// ---------- types ----------
typedef __bf16 bf16_t;
typedef bf16_t v16bf __attribute__((ext_vector_type(16)));
typedef bf16_t v8bf  __attribute__((ext_vector_type(8)));
typedef bf16_t v2bf  __attribute__((ext_vector_type(2)));
typedef float  v8f   __attribute__((ext_vector_type(8)));

union V16 { v16bf v; v8bf h2[2]; };

// Native converts: let clang select v_cvt_(pk_)bf16_f32 instead of bit-twiddling.
static __device__ inline bf16_t f2bf(float f) { return (bf16_t)f; }
static __device__ inline float  bf2f(bf16_t b) { return (float)b; }
static __device__ inline unsigned pk2(float a, float b) {   // two bf16 in one dword
    v2bf v; v[0] = (bf16_t)a; v[1] = (bf16_t)b;
    return __builtin_bit_cast(unsigned, v);
}
static __device__ inline float loadf(const float* p)  { return *p; }
static __device__ inline float loadf(const bf16_t* p) { return bf2f(*p); }
static __device__ inline void  storeo(float* p, float v)  { *p = v; }
static __device__ inline void  storeo(bf16_t* p, float v) { *p = f2bf(v); }

static __device__ inline v8f wmma_bf16(v16bf a, v16bf b, v8f c) {
    return __builtin_amdgcn_wmma_f32_16x16x32_bf16(false, a, false, b, (short)0, c,
                                                   false, false);
}

// ---------- constants ----------
#define Bq   8
#define LQ   32
#define LKV  4096
#define NH   16
#define HD   64
#define HS   1024

// =====================================================================
// Fused K/V projection GEMM:  Ck = A*Wk, Cv = A*Wv  (bf16 outputs)
// block = 256 threads (8 waves), tile 128x128, K-step 32.
// A tile staged ONCE per K-step, reused for both weight matrices.
// =====================================================================
__global__ __launch_bounds__(256)
void gemm_kv(const float* __restrict__ A, const float* __restrict__ Wk,
             const float* __restrict__ Wv, bf16_t* __restrict__ Ck,
             bf16_t* __restrict__ Cv, int M, int N, int K)
{
    __shared__ bf16_t As [128][40];   // row-major A tile (80B rows)
    __shared__ bf16_t Btk[128][40];   // transposed Wk tile: Bt[n][k]
    __shared__ bf16_t Btv[128][40];   // transposed Wv tile

    const int tid  = threadIdx.x;
    const int wave = tid >> 5, lane = tid & 31;
    const int half = lane >> 4, l16 = lane & 15;
    const int m0 = blockIdx.x * 128, n0 = blockIdx.y * 128;
    const int wr = wave >> 2, wc = wave & 3;

    v8f acck[4][2] = {};
    v8f accv[4][2] = {};

    for (int k0 = 0; k0 < K; k0 += 32) {
        // ---- stage A tile: 128x32, packed 8B LDS stores ----
#pragma unroll
        for (int it = 0; it < 4; ++it) {
            int e = tid * 4 + it * 1024;
            int r = e >> 5, c = e & 31;
            const float* ap = A + (size_t)(m0 + r) * K + k0 + c;
            if (k0 + 32 < K) __builtin_prefetch(ap + 32, 0, 3);
            float4 av = *(const float4*)ap;
            uint2 pk; pk.x = pk2(av.x, av.y); pk.y = pk2(av.z, av.w);
            *(uint2*)&As[r][c] = pk;
        }
        // ---- stage both W tiles transposed: column loads + packed stores ----
#pragma unroll
        for (int it = 0; it < 4; ++it) {
            int slot = tid + it * 256;            // 1024 slots = 128 n x 8 kgroups
            int c = slot & 127, kg = slot >> 7;
            const float* wpk = Wk + (size_t)(k0 + kg * 4) * N + n0 + c;
            const float* wpv = Wv + (size_t)(k0 + kg * 4) * N + n0 + c;
            if (k0 + 32 < K) {
                __builtin_prefetch(wpk + (size_t)32 * N, 0, 3);
                __builtin_prefetch(wpv + (size_t)32 * N, 0, 3);
            }
            uint2 pk;
            pk.x = pk2(wpk[0], wpk[(size_t)N]);
            pk.y = pk2(wpk[(size_t)2 * N], wpk[(size_t)3 * N]);
            *(uint2*)&Btk[c][kg * 4] = pk;
            pk.x = pk2(wpv[0], wpv[(size_t)N]);
            pk.y = pk2(wpv[(size_t)2 * N], wpv[(size_t)3 * N]);
            *(uint2*)&Btv[c][kg * 4] = pk;
        }
        __syncthreads();

        // ---- WMMA compute: 16 wmma per K-step per wave ----
        V16 bk[2], bv[2];
#pragma unroll
        for (int j = 0; j < 2; ++j) {
            const bf16_t* p = &Btk[wc * 32 + j * 16 + l16][half * 16];
            bk[j].h2[0] = *(const v8bf*)p;  bk[j].h2[1] = *(const v8bf*)(p + 8);
            const bf16_t* q = &Btv[wc * 32 + j * 16 + l16][half * 16];
            bv[j].h2[0] = *(const v8bf*)q;  bv[j].h2[1] = *(const v8bf*)(q + 8);
        }
#pragma unroll
        for (int i = 0; i < 4; ++i) {
            const bf16_t* p = &As[wr * 64 + i * 16 + l16][half * 16];
            V16 a;
            a.h2[0] = *(const v8bf*)p;
            a.h2[1] = *(const v8bf*)(p + 8);
#pragma unroll
            for (int j = 0; j < 2; ++j) {
                acck[i][j] = wmma_bf16(a.v, bk[j].v, acck[i][j]);
                accv[i][j] = wmma_bf16(a.v, bv[j].v, accv[i][j]);
            }
        }
        __syncthreads();
    }

    // ---- epilogue ----
#pragma unroll
    for (int i = 0; i < 4; ++i)
#pragma unroll
        for (int j = 0; j < 2; ++j)
#pragma unroll
            for (int v = 0; v < 8; ++v) {
                int r = m0 + wr * 64 + i * 16 + v + half * 8;
                int c = n0 + wc * 32 + j * 16 + l16;
                Ck[(size_t)r * N + c] = f2bf(acck[i][j][v]);
                Cv[(size_t)r * N + c] = f2bf(accv[i][j][v]);
            }
}

// =====================================================================
// Generic tiled WMMA GEMM:  C[M,N] = A[M,K] * W[K,N]
// (used for the small Q and Wo GEMMs)
// =====================================================================
template <typename TIN, typename TOUT>
__global__ __launch_bounds__(256)
void gemm_wmma(const TIN* __restrict__ A, const float* __restrict__ W,
               TOUT* __restrict__ C, int M, int N, int K)
{
    __shared__ bf16_t As[128][40];
    __shared__ bf16_t Bt[128][40];

    const int tid  = threadIdx.x;
    const int wave = tid >> 5, lane = tid & 31;
    const int half = lane >> 4, l16 = lane & 15;
    const int m0 = blockIdx.x * 128, n0 = blockIdx.y * 128;
    const int wr = wave >> 2, wc = wave & 3;

    v8f acc[4][2] = {};

    for (int k0 = 0; k0 < K; k0 += 32) {
#pragma unroll
        for (int it = 0; it < 4; ++it) {
            int e = tid * 4 + it * 1024;
            int r = e >> 5, c = e & 31;
            const TIN* ap = A + (size_t)(m0 + r) * K + k0 + c;
            if (k0 + 32 < K) __builtin_prefetch(ap + 32, 0, 3);
            float t0 = loadf(ap), t1 = loadf(ap + 1);
            float t2 = loadf(ap + 2), t3 = loadf(ap + 3);
            uint2 pk; pk.x = pk2(t0, t1); pk.y = pk2(t2, t3);
            *(uint2*)&As[r][c] = pk;
        }
#pragma unroll
        for (int it = 0; it < 4; ++it) {
            int slot = tid + it * 256;
            int c = slot & 127, kg = slot >> 7;
            const float* wp = W + (size_t)(k0 + kg * 4) * N + n0 + c;
            if (k0 + 32 < K) __builtin_prefetch(wp + (size_t)32 * N, 0, 3);
            uint2 pk;
            pk.x = pk2(wp[0], wp[(size_t)N]);
            pk.y = pk2(wp[(size_t)2 * N], wp[(size_t)3 * N]);
            *(uint2*)&Bt[c][kg * 4] = pk;
        }
        __syncthreads();

        V16 bfrag[2];
#pragma unroll
        for (int j = 0; j < 2; ++j) {
            const bf16_t* p = &Bt[wc * 32 + j * 16 + l16][half * 16];
            bfrag[j].h2[0] = *(const v8bf*)p;
            bfrag[j].h2[1] = *(const v8bf*)(p + 8);
        }
#pragma unroll
        for (int i = 0; i < 4; ++i) {
            const bf16_t* p = &As[wr * 64 + i * 16 + l16][half * 16];
            V16 a;
            a.h2[0] = *(const v8bf*)p;
            a.h2[1] = *(const v8bf*)(p + 8);
#pragma unroll
            for (int j = 0; j < 2; ++j)
                acc[i][j] = wmma_bf16(a.v, bfrag[j].v, acc[i][j]);
        }
        __syncthreads();
    }

#pragma unroll
    for (int i = 0; i < 4; ++i)
#pragma unroll
        for (int j = 0; j < 2; ++j)
#pragma unroll
            for (int v = 0; v < 8; ++v) {
                int r = m0 + wr * 64 + i * 16 + v + half * 8;
                int c = n0 + wc * 32 + j * 16 + l16;
                storeo(C + (size_t)r * N + c, acc[i][j][v]);
            }
}

// =====================================================================
// RoPE (query only, fp32 trig) -> Qbf [B][H][LQ][D] bf16
// =====================================================================
__global__ __launch_bounds__(256)
void rope_kernel(const float* __restrict__ Qf, const long long* __restrict__ pos,
                 bf16_t* __restrict__ Qbf)
{
    int idx = blockIdx.x * 256 + threadIdx.x;     // over B*H*LQ*D = 262144
    int d = idx & 63;
    int q = (idx >> 6) & 31;
    int h = (idx >> 11) & 15;
    int b = idx >> 15;
    float p = (float)pos[b * LQ + q];
    int fi = d & 31;
    float inv  = __expf(-(float)fi * (1.0f / 32.0f) * 9.210340371976184f); // ln(1e4)
    float ang  = p * inv;
    float c = cosf(ang), s = sinf(ang);
    const float* qrow = Qf + ((size_t)(b * LQ + q)) * HS + h * HD;
    float x = qrow[d];
    float other = (d < 32) ? -qrow[d + 32] : qrow[d - 32];
    Qbf[idx] = f2bf(x * c + other * s);
}

// =====================================================================
// Flash-style attention: one block per (b,h). 8 waves, 128-token chunks.
// K fragments software-pipelined: next chunk's loads issue before softmax.
// =====================================================================
__global__ __launch_bounds__(256)
void attn_kernel(const bf16_t* __restrict__ Qbf, const bf16_t* __restrict__ Kbf,
                 const bf16_t* __restrict__ Vbf, bf16_t* __restrict__ Obf)
{
    __shared__ float  Sc[32][133];       // fp32 scores chunk (odd stride)
    __shared__ bf16_t Pb[32][136];       // bf16 probabilities (A-frag source)
    __shared__ bf16_t Vt[64][136];       // V chunk transposed: Vt[d][token]
    __shared__ float  red[8][32];
    __shared__ float  m_row[32], l_row[32], corr[32];

    const int tid  = threadIdx.x;
    const int wave = tid >> 5, lane = tid & 31;
    const int half = lane >> 4, l16 = lane & 15;
    const int b = blockIdx.x >> 4, h = blockIdx.x & 15;

    if (tid < 32) { m_row[tid] = -1e30f; l_row[tid] = 0.0f; }

    // Q fragments: [mt][kc], A layout, loaded once (global is d-contiguous)
    V16 qf[2][2];
    const bf16_t* Qbase = Qbf + ((size_t)(b * NH + h)) * LQ * HD;
#pragma unroll
    for (int mt = 0; mt < 2; ++mt)
#pragma unroll
        for (int kc = 0; kc < 2; ++kc) {
            const bf16_t* r = Qbase + (mt * 16 + l16) * HD + kc * 32 + half * 16;
            qf[mt][kc].h2[0] = *(const v8bf*)r;
            qf[mt][kc].h2[1] = *(const v8bf*)(r + 8);
        }

    const int mt_o = wave >> 2, nt_o = wave & 3;   // O tile role of this wave
    v8f oacc = {};

    const bf16_t* Kb = Kbf + (size_t)b * LKV * HS + h * HD;
    const bf16_t* Vb = Vbf + (size_t)b * LKV * HS + h * HD;

    // preload K fragments for chunk 0
    V16 kf[2];
#pragma unroll
    for (int kc = 0; kc < 2; ++kc) {
        const bf16_t* kp = Kb + (size_t)(wave * 16 + l16) * HS + kc * 32 + half * 16;
        kf[kc].h2[0] = *(const v8bf*)kp;
        kf[kc].h2[1] = *(const v8bf*)(kp + 8);
    }

    __syncthreads();

    for (int kv0 = 0; kv0 < LKV; kv0 += 128) {
        if (kv0 + 128 < LKV) {
            __builtin_prefetch(Kb + (size_t)(kv0 + 128 + wave * 16 + l16) * HS, 0, 3);
            __builtin_prefetch(Vb + (size_t)(kv0 + 128 + wave * 16 + l16) * HS, 0, 3);
        }
        // ---- P1: S = (Q K^T)/8 for this wave's 16 columns ----
#pragma unroll
        for (int mt = 0; mt < 2; ++mt) {
            v8f s = {};
#pragma unroll
            for (int kc = 0; kc < 2; ++kc)
                s = wmma_bf16(qf[mt][kc].v, kf[kc].v, s);
#pragma unroll
            for (int v = 0; v < 8; ++v)
                Sc[mt * 16 + v + half * 8][wave * 16 + l16] = s[v] * 0.125f;
        }
        // issue next chunk's K loads now; latency overlaps the softmax phases
        if (kv0 + 128 < LKV) {
#pragma unroll
            for (int kc = 0; kc < 2; ++kc) {
                const bf16_t* kp = Kb + (size_t)(kv0 + 128 + wave * 16 + l16) * HS
                                      + kc * 32 + half * 16;
                kf[kc].h2[0] = *(const v8bf*)kp;
                kf[kc].h2[1] = *(const v8bf*)(kp + 8);
            }
        }
        __syncthreads();

        // ---- P2: online softmax over the 128-column chunk ----
        const int r = tid & 31, seg = tid >> 5;
        float mx = -1e30f;
#pragma unroll
        for (int j = 0; j < 16; ++j) mx = fmaxf(mx, Sc[r][seg * 16 + j]);
        red[seg][r] = mx;
        __syncthreads();
        if (tid < 32) {
            float mo = m_row[tid], mn = mo;
#pragma unroll
            for (int s2 = 0; s2 < 8; ++s2) mn = fmaxf(mn, red[s2][tid]);
            m_row[tid] = mn;
            corr[tid] = __expf(mo - mn);
        }
        __syncthreads();
        {
            float mn = m_row[r], psum = 0.0f;
#pragma unroll
            for (int j = 0; j < 16; ++j) {
                float p = __expf(Sc[r][seg * 16 + j] - mn);
                Pb[r][seg * 16 + j] = f2bf(p);
                psum += p;
            }
            red[seg][r] = psum;
        }
        __syncthreads();
        if (tid < 32) {
            float acc = l_row[tid] * corr[tid];
#pragma unroll
            for (int s2 = 0; s2 < 8; ++s2) acc += red[s2][tid];
            l_row[tid] = acc;
        }
        // ---- P3: stage V chunk transposed ----
#pragma unroll
        for (int it = 0; it < 4; ++it) {
            int slot = tid + it * 256;               // 1024 slots = 128 tok x 8 grp
            int tl = slot & 127, dg = slot >> 7;
            v8bf vv = *(const v8bf*)(Vb + (size_t)(kv0 + tl) * HS + dg * 8);
#pragma unroll
            for (int j = 0; j < 8; ++j) Vt[dg * 8 + j][tl] = vv[j];
        }
        __syncthreads();

        // ---- P4: O = O*corr + P V ----
#pragma unroll
        for (int v = 0; v < 8; ++v)
            oacc[v] *= corr[mt_o * 16 + v + half * 8];
#pragma unroll
        for (int kc = 0; kc < 4; ++kc) {
            const bf16_t* pp = &Pb[mt_o * 16 + l16][kc * 32 + half * 16];
            V16 pf;
            pf.h2[0] = *(const v8bf*)pp;
            pf.h2[1] = *(const v8bf*)(pp + 8);
            const bf16_t* vp = &Vt[nt_o * 16 + l16][kc * 32 + half * 16];
            V16 vf;
            vf.h2[0] = *(const v8bf*)vp;
            vf.h2[1] = *(const v8bf*)(vp + 8);
            oacc = wmma_bf16(pf.v, vf.v, oacc);
        }
        __syncthreads();
    }

    // ---- epilogue: normalize and store bf16 [B][LQ][HS] ----
#pragma unroll
    for (int v = 0; v < 8; ++v) {
        int q = mt_o * 16 + v + half * 8;
        float val = oacc[v] / l_row[q];
        Obf[((size_t)(b * LQ + q)) * HS + h * HD + nt_o * 16 + l16] = f2bf(val);
    }
}

// =====================================================================
// launcher
// =====================================================================
extern "C" void kernel_launch(void* const* d_in, const int* in_sizes, int n_in,
                              void* d_out, int out_size, void* d_ws, size_t ws_size,
                              hipStream_t stream)
{
    const float*     hs  = (const float*)d_in[0];
    const float*     ehs = (const float*)d_in[1];
    const long long* pos = (const long long*)d_in[2];
    const float*     Wq  = (const float*)d_in[3];
    const float*     Wk  = (const float*)d_in[4];
    const float*     Wv  = (const float*)d_in[5];
    const float*     Wo  = (const float*)d_in[6];
    float* out = (float*)d_out;

    char* ws = (char*)d_ws;
    const size_t KV_BYTES = (size_t)Bq * LKV * HS * sizeof(bf16_t);   // 64 MB
    bf16_t* Kbf = (bf16_t*)(ws);
    bf16_t* Vbf = (bf16_t*)(ws + KV_BYTES);
    float*  Qf  = (float*)(ws + 2 * KV_BYTES);                        // 1 MB
    bf16_t* Qbf = (bf16_t*)(ws + 2 * KV_BYTES + (size_t)Bq * LQ * HS * 4);
    bf16_t* Obf = (bf16_t*)(ws + 2 * KV_BYTES + (size_t)Bq * LQ * HS * 4
                               + (size_t)Bq * LQ * HS * 2);

    dim3 blk(256);
    // fused K+V projections: [32768,1024] x [1024,1024] x2 -> bf16
    gemm_kv<<<dim3(256, 8), blk, 0, stream>>>(ehs, Wk, Wv, Kbf, Vbf,
                                              Bq * LKV, HS, HS);
    // Q projection: [256,1024] -> fp32 (RoPE precision)
    gemm_wmma<float, float><<<dim3(2, 8), blk, 0, stream>>>(hs, Wq, Qf,
                                                            Bq * LQ, HS, HS);
    // RoPE -> Qbf [B][H][LQ][D]
    rope_kernel<<<dim3((Bq * NH * LQ * HD) / 256), blk, 0, stream>>>(Qf, pos, Qbf);
    // attention -> Obf [B][LQ][HS]
    attn_kernel<<<dim3(Bq * NH), blk, 0, stream>>>(Qbf, Kbf, Vbf, Obf);
    // output projection -> fp32 d_out
    gemm_wmma<bf16_t, float><<<dim3(2, 8), blk, 0, stream>>>(Obf, Wo, out,
                                                             Bq * LQ, HS, HS);
}
